// MultiHeadAttentionLayer_11063835754631
// MI455X (gfx1250) — matrix operand order repaired and checked by reading, not compiled, not run on hardware
//
#include <hip/hip_runtime.h>
#include <hip/hip_bf16.h>

typedef _Float16 v16h __attribute__((ext_vector_type(16)));
typedef _Float16 v8h  __attribute__((ext_vector_type(8)));
typedef float    v8f  __attribute__((ext_vector_type(8)));
typedef unsigned int u32x4 __attribute__((ext_vector_type(4)));
typedef int          i32x8 __attribute__((ext_vector_type(8)));
typedef int          i32x4 __attribute__((ext_vector_type(4)));

#define AS1 __attribute__((address_space(1)))
#define AS3 __attribute__((address_space(3)))

#define NN   16000
#define EE   256000
#define DIN  256
#define HD   256   // H*D
#define NH   8

#if __has_builtin(__builtin_amdgcn_tensor_load_to_lds)
#define USE_TDM 1
#elif __has_builtin(__builtin_amdgcn_global_load_async_to_lds_b128)
#define USE_ASYNC 1
#endif

static __device__ __forceinline__ v16h cat8(v8h a, v8h b) {
  return __builtin_shufflevector(a, b, 0,1,2,3,4,5,6,7,8,9,10,11,12,13,14,15);
}

static __device__ __forceinline__ v8h cvt8(v8f f) {
  v8h h;
#pragma unroll
  for (int i = 0; i < 8; ++i) h[i] = (_Float16)f[i];
  return h;
}

__global__ __launch_bounds__(256) void k_zero(float* __restrict__ p, int n) {
  int i = blockIdx.x * 256 + threadIdx.x;
  if (i < n) p[i] = 0.f;
}

__global__ __launch_bounds__(256) void k_pack_x(const float* __restrict__ x,
                                                _Float16* __restrict__ xh) {
  int i = blockIdx.x * 256 + threadIdx.x;
  if (i < NN * DIN) xh[i] = (_Float16)x[i];
}

// wt[sel][n*256 + k] = W_sel[k][n]  (transposed, f16). We: rows 0..255 only.
__global__ __launch_bounds__(256) void k_pack_w(const float* __restrict__ Wq,
                                                const float* __restrict__ Wk,
                                                const float* __restrict__ Wv,
                                                const float* __restrict__ We,
                                                _Float16* __restrict__ wt) {
  int i = blockIdx.x * 256 + threadIdx.x;
  if (i >= 4 * DIN * HD) return;
  int sel = i >> 16;
  int r   = i & 65535;
  int n   = r >> 8;    // output column
  int k   = r & 255;   // input feature
  const float* W = (sel == 0) ? Wq : (sel == 1) ? Wk : (sel == 2) ? Wv : We;
  wt[i] = (_Float16)W[k * HD + n];
}

// One wave computes one 16x16 f32 tile of {q,k,v}; grid.y: sel(3) x ntile(16).
__global__ __launch_bounds__(32) void k_gemm_qkv(const _Float16* __restrict__ xh,
                                                 const _Float16* __restrict__ wt,
                                                 float* __restrict__ qkv) {
  int m0  = blockIdx.x * 16;
  int by  = blockIdx.y;
  int sel = by >> 4;
  int nt  = by & 15;
  const _Float16* w = wt + (size_t)sel * (DIN * HD);
  float* out = qkv + (size_t)sel * ((size_t)NN * HD);

  int lane = threadIdx.x;
  int lm = lane & 15;
  int hi = lane >> 4;
  int n  = nt * 16 + lm;

  const _Float16* ap = xh + (size_t)(m0 + lm) * DIN + hi * 16;  // A: 16 contiguous K
  const _Float16* bp = w  + (size_t)n * DIN + hi * 16;          // B: 16 contiguous K

  v8f c = {};
#pragma unroll
  for (int k0 = 0; k0 < DIN; k0 += 32) {
    v8h a0 = *(const v8h*)(ap + k0);
    v8h a1 = *(const v8h*)(ap + k0 + 8);
    v16h A = cat8(a0, a1);
    v16h B = *(const v16h*)(bp + k0);
    c = __builtin_amdgcn_wmma_f32_16x16x32_f16(false, A, false, B, (short)0, c,
                                               false, false);
  }
#pragma unroll
  for (int i = 0; i < 8; ++i)
    out[(size_t)(m0 + i + hi * 8) * HD + n] = c[i];
}

// 512 threads = 16 waves; one 16-edge tile per block; wave w owns columns w*16..+15.
__global__ __launch_bounds__(512) void k_edge(
    const float* __restrict__ edge_attr,
    const int*   __restrict__ eidx,
    const float* __restrict__ coords,
    const _Float16* __restrict__ we_t,
    const float* __restrict__ We,       // f32 original, for last row (dist column)
    const float* __restrict__ qws,
    const float* __restrict__ kws,
    const float* __restrict__ vws,
    float* __restrict__ e_out,
    float* __restrict__ wV,
    float* __restrict__ zacc) {
  __shared__ float sAf[16 * 264];       // f32 tile, 264-dword row stride (8-dword pad)
  __shared__ int   s_row[16], s_col[16];
  __shared__ float s_dist[16];
  __shared__ float s_asum[16][8];
  __shared__ float s_ax[16][8];

  int tid = threadIdx.x;
  int e0  = blockIdx.x * 16;

  // ---- stage edge_attr tile (16 x 256 f32) into LDS ----
#if defined(USE_TDM)
  if (tid < 32) {   // TDM is wave-scoped (EXEC ignored): issue from wave 0 only
    unsigned long long ga = (unsigned long long)(edge_attr + (size_t)e0 * DIN);
    unsigned ldsoff = (unsigned)(size_t)(AS3 float*)sAf;
    // D# group0: count=1 | lds_addr | global_addr[56:0] | type=2 (bits 127:126)
    u32x4 g0 = { 1u, ldsoff, (unsigned)ga, (unsigned)(ga >> 32) | 0x80000000u };
    // D# group1: data_size=4B, pad_enable, pad_interval=256dw(7), pad_amount=8dw(7);
    // tensor_dim0=256, tensor_dim1=16, tile_dim0=256, tile_dim1=16, stride0=256.
    i32x8 g1 = { 0x0FD20000, 0x01000000, 0x00100000, 0x01000000, 16, 256, 0, 0 };
    i32x4 gz = { 0, 0, 0, 0 };          // groups 2/3 unused (2-D tile)
    i32x8 gz8 = { 0, 0, 0, 0, 0, 0, 0, 0 };
    __builtin_amdgcn_tensor_load_to_lds(g0, g1, gz, gz, gz8, 0);
    __builtin_amdgcn_s_wait_tensorcnt(0);
  }
#elif defined(USE_ASYNC)
  for (int i = tid; i < 1024; i += 512) {        // 1024 x b128 = 16 KB
    int m = i >> 6, kk = (i & 63) << 2;
    __builtin_amdgcn_global_load_async_to_lds_b128(
        (const AS1 void*)(edge_attr + (size_t)(e0 + m) * DIN + kk),
        (AS3 void*)&sAf[m * 264 + kk], 0, 0);
  }
#if __has_builtin(__builtin_amdgcn_s_wait_asynccnt)
  __builtin_amdgcn_s_wait_asynccnt(0);
#else
  asm volatile("s_wait_asynccnt 0x0" ::: "memory");
#endif
#else
  for (int i = tid; i < 16 * 256; i += 512) {
    int m = i >> 8, kk = i & 255;
    sAf[m * 264 + kk] = edge_attr[(size_t)(e0 + m) * DIN + kk];
  }
#endif

  if (tid < 16) {
    int e  = e0 + tid;
    int r  = eidx[e];
    int cc = eidx[EE + e];
    s_row[tid] = r;
    s_col[tid] = cc;
    float dx = coords[r * 3 + 0] - coords[cc * 3 + 0];
    float dy = coords[r * 3 + 1] - coords[cc * 3 + 1];
    float dz = coords[r * 3 + 2] - coords[cc * 3 + 2];
    s_dist[tid] = 0.1f * sqrtf(dx * dx + dy * dy + dz * dz);
  }
  if (tid < 128) s_asum[tid >> 3][tid & 7] = 0.f;
  // Prefetch next tile's edge_attr into L2 (covers 16 KB in 128B strides).
  if (tid < 128 && e0 + 16 < EE)
    __builtin_prefetch(edge_attr + (size_t)(e0 + 16) * DIN + tid * 32, 0, 1);
  __syncthreads();

  int wv   = tid >> 5;
  int lane = tid & 31;
  int lm   = lane & 15;
  int hi   = lane >> 4;
  int n    = wv * 16 + lm;     // global output column 0..255
  int head = wv >> 1;          // whole wave sits in one head

  const float*    apf = sAf + lm * 264 + hi * 16;
  const _Float16* bp  = we_t + (size_t)n * DIN + hi * 16;

  v8f c = {};
#pragma unroll
  for (int k0 = 0; k0 < DIN; k0 += 32) {
    v8f f0 = *(const v8f*)(apf + k0);
    v8f f1 = *(const v8f*)(apf + k0 + 8);
    v16h A = cat8(cvt8(f0), cvt8(f1));
    v16h B = *(const v16h*)(bp + k0);
    c = __builtin_amdgcn_wmma_f32_16x16x32_f16(false, A, false, B, (short)0, c,
                                               false, false);
  }

  float we_last = We[256 * 256 + n];          // last row of We (dist feature), f32
  const float rs = 0.17677669529663689f;      // 1/sqrt(32)

#pragma unroll
  for (int i = 0; i < 8; ++i) {
    int m = i + hi * 8;                       // edge index within tile (C layout)
    float ea = c[i] + s_dist[m] * we_last;    // exact K=257 term in f32
    int rn = s_row[m], cn = s_col[m];
    float kq = kws[(size_t)rn * HD + n] * qws[(size_t)cn * HD + n] * rs;
    kq = fminf(5.f, fmaxf(-5.f, kq));
    float alpha = kq * ea;
    e_out[(size_t)(e0 + m) * HD + n] = alpha;
    // Sum alpha over this wave's 16 columns (stays within its 16-lane half).
    float s = alpha;
    s += __shfl_xor(s, 1);
    s += __shfl_xor(s, 2);
    s += __shfl_xor(s, 4);
    s += __shfl_xor(s, 8);
    if (lm == 0) atomicAdd(&s_asum[m][head], s);   // two waves per head combine
  }
  __syncthreads();

  if (tid < 128) {
    int m = tid >> 3, h = tid & 7;
    float ax = __expf(fminf(5.f, fmaxf(-5.f, s_asum[m][h])));
    s_ax[m][h] = ax;
    atomicAdd(&zacc[(size_t)s_col[m] * NH + h], ax);
  }
  __syncthreads();

#pragma unroll
  for (int i = 0; i < 8; ++i) {
    int m = i + hi * 8;
    float val = vws[(size_t)s_row[m] * HD + n] * s_ax[m][head];
    atomicAdd(&wV[(size_t)s_col[m] * HD + n], val);
  }
}

__global__ __launch_bounds__(256) void k_final(const float* __restrict__ wV,
                                               const float* __restrict__ zacc,
                                               float* __restrict__ h_out) {
  int i = blockIdx.x * 256 + threadIdx.x;
  if (i >= NN * HD) return;
  int node = i >> 8;
  int h    = (i & 255) >> 5;
  h_out[i] = wV[i] / (zacc[node * NH + h] + 1e-6f);
}

__global__ __launch_bounds__(256) void k_coords(const float* __restrict__ coords,
                                                float* __restrict__ outc) {
  int i = blockIdx.x * 256 + threadIdx.x;
  if (i < NN * 3) outc[i] = coords[i];
}

extern "C" void kernel_launch(void* const* d_in, const int* in_sizes, int n_in,
                              void* d_out, int out_size, void* d_ws, size_t ws_size,
                              hipStream_t stream) {
  (void)in_sizes; (void)n_in; (void)out_size; (void)ws_size;
  const float* x         = (const float*)d_in[0];
  const float* edge_attr = (const float*)d_in[1];
  const int*   eidx      = (const int*)d_in[2];
  const float* coords    = (const float*)d_in[3];
  const float* Wq        = (const float*)d_in[4];
  const float* Wk        = (const float*)d_in[5];
  const float* Wv        = (const float*)d_in[6];
  const float* We        = (const float*)d_in[7];

  // Workspace layout (bytes):
  //   0        : x_h16    (16000*256 f16)  =  8,192,000
  //   8192000  : wt (Wq,Wk,Wv,We transposed f16, 4*65536) = 524,288
  //   8716288  : q,k,v f32 (3 * 16,384,000)
  //   57868288 : wV f32   (16,384,000)
  //   74252288 : z  f32   (512,000)          -> total ~74.8 MB
  char* ws = (char*)d_ws;
  _Float16* xh = (_Float16*)(ws + 0);
  _Float16* wt = (_Float16*)(ws + 8192000);
  float* qkv = (float*)(ws + 8716288);
  float* q = qkv;
  float* k = qkv + 4096000;
  float* v = qkv + 8192000;
  float* wV = (float*)(ws + 57868288);
  float* z  = (float*)(ws + 74252288);

  float* h_out = (float*)d_out;            //  4,096,000 floats
  float* e_out = h_out + 4096000;          // 65,536,000 floats
  float* c_out = h_out + 69632000;         //     48,000 floats

  k_zero<<<(4224000 + 255) / 256, 256, 0, stream>>>(wV, 4224000);  // wV + z contiguous
  k_pack_x<<<(NN * DIN + 255) / 256, 256, 0, stream>>>(x, xh);
  k_pack_w<<<(4 * 65536 + 255) / 256, 256, 0, stream>>>(Wq, Wk, Wv, We, wt);
  k_gemm_qkv<<<dim3(NN / 16, 48), 32, 0, stream>>>(xh, wt, qkv);
  k_edge<<<EE / 16, 512, 0, stream>>>(edge_attr, eidx, coords, wt + 3 * 65536, We,
                                      q, k, v, e_out, wV, z);
  k_final<<<(NN * HD + 255) / 256, 256, 0, stream>>>(wV, z, h_out);
  k_coords<<<(NN * 3 + 255) / 256, 256, 0, stream>>>(coords, c_out);
}
